// ProperKVMemoryBlock_44547400794258
// MI455X (gfx1250) — compile-verified, matrix operand order checked by reading, not run on hardware
//
#include <hip/hip_runtime.h>
#include <hip/hip_bf16.h>
#include <math.h>

#define BB 2
#define LL 2048
#define DD 256
#define PP 32
#define BLT (BB*LL)      // 4096 rows total
#define CH 64            // chunk length
#define NC (BLT/CH)      // 64 chunks total
#define FEAT 64          // 2P phasor features

typedef __attribute__((ext_vector_type(16))) _Float16 v16h;
typedef __attribute__((ext_vector_type(8)))  float    v8f;

// ---------------- WMMA helpers (gfx1250 wave32, 16x16x32 f16 -> f32) --------

// Load a 16x32 f16 A-fragment (or 32x16 B-fragment, symmetric layout).
// Element (i,k) lives at base[i*rs + k*cs].  For A: i=row(M), for B: i=col(N).
// With cs==1 each lane reads two contiguous 16B runs -> global_load_b128 x2.
__device__ __forceinline__ v16h ld_frag(const _Float16* base, int lane,
                                        long rs, long cs) {
  int  i    = lane & 15;
  long koff = (lane & 16) ? 8 : 0;
  const _Float16* p = base + (long)i * rs;
  v16h f;
#pragma unroll
  for (int v = 0; v < 8; ++v) {
    long kk = 2 * v + ((v >= 4) ? 8 : 0) + koff;
    f[2 * v]     = p[kk * cs];
    f[2 * v + 1] = p[(kk + 1) * cs];
  }
  return f;
}

__device__ __forceinline__ v8f wmma_f16(v16h a, v16h b, v8f c) {
  return __builtin_amdgcn_wmma_f32_16x16x32_f16(false, a, false, b,
                                                (short)0, c, false, false);
}

__device__ __forceinline__ float gelu_exact(float x) {
  return 0.5f * x * (1.0f + erff(x * 0.70710678118654752f));
}

// ---------------- f32 -> f16 conversions ------------------------------------

__global__ __launch_bounds__(256) void cvt_block(const float* __restrict__ src,
                                                 _Float16* __restrict__ dst,
                                                 int rows, int cols, int ldd) {
  int i = blockIdx.x * 256 + threadIdx.x;
  if (i >= rows * cols) return;
  int r = i / cols, c = i % cols;
  dst[(size_t)r * ldd + c] = (_Float16)src[i];
}

// transpose + convert: src row-major [rows, cols] -> dst[c*ldd + r] (f16)
__global__ __launch_bounds__(256) void cvt_t(const float* __restrict__ src,
                                             _Float16* __restrict__ dst,
                                             int rows, int cols, int ldd) {
  int i = blockIdx.x * 256 + threadIdx.x;
  if (i >= rows * cols) return;
  int r = i / cols, c = i % cols;
  dst[(size_t)c * ldd + r] = (_Float16)src[i];
}

// ---------------- generic f16 GEMM: C[M,N] = A[M,K] x BT[N,K]ᵀ --------------
// Wave computes a 16x32 output slab (two 16x16 WMMA tiles sharing an A-frag).

__global__ __launch_bounds__(256) void gemm_f16f32(const _Float16* __restrict__ A,
                                                   const _Float16* __restrict__ BT,
                                                   float* __restrict__ C,
                                                   int M, int N, int K) {
  int gw   = blockIdx.x * 8 + (threadIdx.x >> 5);
  int lane = threadIdx.x & 31;
  int nGroups = N >> 5;
  int mt = gw / nGroups;
  int ng = gw % nGroups;
  if (mt >= (M >> 4)) return;
  v8f acc0 = {}, acc1 = {};
  const _Float16* Abase = A + (size_t)mt * 16 * K;
  const _Float16* B0    = BT + (size_t)(ng * 32) * K;       // rows n, contiguous K
  const _Float16* B1    = BT + (size_t)(ng * 32 + 16) * K;
  for (int k = 0; k < K; k += 32) {
    __builtin_prefetch(Abase + k + 64, 0, 1);               // global_prefetch_b8
    __builtin_prefetch(B0 + k + 64, 0, 1);
    v16h af = ld_frag(Abase + k, lane, K, 1);
    v16h b0 = ld_frag(B0 + k, lane, K, 1);
    v16h b1 = ld_frag(B1 + k, lane, K, 1);
    acc0 = wmma_f16(af, b0, acc0);
    acc1 = wmma_f16(af, b1, acc1);
  }
  int n = lane & 15, ro = (lane & 16) ? 8 : 0;
#pragma unroll
  for (int v = 0; v < 8; ++v) {
    size_t r = (size_t)(mt * 16 + v + ro);
    C[r * N + ng * 32 + n]      = acc0[v];
    C[r * N + ng * 32 + 16 + n] = acc1[v];
  }
}

// ---------------- epilogues -------------------------------------------------

// phases -> phasor features Q = [cos | sin], f16 (row-major, A-side use)
__global__ __launch_bounds__(32) void phase_kernel(const float* __restrict__ Y1,
                                                   const float* __restrict__ bk,
                                                   _Float16* __restrict__ Q,
                                                   int ldY) {
  int t = blockIdx.x, p = threadIdx.x;
  float y  = Y1[(size_t)t * ldY + p] + bk[p];
  float ph = tanhf(y) * 3.14159265358979323846f;
  Q[(size_t)t * FEAT + p]      = (_Float16)cosf(ph);
  Q[(size_t)t * FEAT + 32 + p] = (_Float16)sinf(ph);
}

// gate scalar g[t]; gated values vgT[d][t] (f16); shifted phasors
// Ksh[t][f] (row-major, B of score GEMM) and KshT[f][t] (A of state GEMM)
__global__ __launch_bounds__(256) void gate_kernel(const float* __restrict__ Y1,
                                                   const float* __restrict__ bv,
                                                   const float* __restrict__ bg1,
                                                   const float* __restrict__ Wg2,
                                                   const float* __restrict__ bg2,
                                                   const _Float16* __restrict__ Q,
                                                   float* __restrict__ g,
                                                   _Float16* __restrict__ vgT,
                                                   _Float16* __restrict__ Ksh,
                                                   _Float16* __restrict__ KshT,
                                                   int ldY) {
  int t = blockIdx.x, d = threadIdx.x;
  int tl = t & (LL - 1);
  float u   = Y1[(size_t)t * ldY + 288 + d];
  float wsh = tl ? Y1[(size_t)(t - 1) * ldY + 544 + d] : 0.f;
  float ge  = gelu_exact(u + wsh + bg1[d]);
  __shared__ float red[256];
  __shared__ float gsh;
  red[d] = ge * Wg2[d];
  __syncthreads();
#pragma unroll
  for (int s = 128; s > 0; s >>= 1) {
    if (d < s) red[d] += red[d + s];
    __syncthreads();
  }
  if (d == 0) {
    float gv = 1.f / (1.f + expf(-(red[0] + bg2[0])));
    gsh = gv;
    g[t] = gv;
  }
  __syncthreads();
  float gv = gsh;
  float val = Y1[(size_t)t * ldY + 32 + d] + bv[d];
  vgT[(size_t)d * BLT + t] = (_Float16)(val * gv);
  if (d < FEAT) {
    _Float16 kv = tl ? Q[(size_t)(t - 1) * FEAT + d] : (_Float16)0.f;
    Ksh[(size_t)t * FEAT + d]  = kv;
    KshT[(size_t)d * BLT + t] = kv;
  }
}

// per-batch inclusive cumsum of g, clipped at 1.0
__global__ __launch_bounds__(256) void gcum_kernel(const float* __restrict__ g,
                                                   float* __restrict__ gc) {
  int b = blockIdx.x, t = threadIdx.x;
  const float* gp = g + (size_t)b * LL;
  float* op = gc + (size_t)b * LL;
  float loc[8];
  float s = 0.f;
#pragma unroll
  for (int j = 0; j < 8; ++j) { loc[j] = gp[t * 8 + j]; s += loc[j]; }
  __shared__ float part[256];
  part[t] = s;
  __syncthreads();
  for (int st = 1; st < 256; st <<= 1) {
    float v = (t >= st) ? part[t - st] : 0.f;
    __syncthreads();
    part[t] += v;
    __syncthreads();
  }
  float run = part[t] - s;  // exclusive
#pragma unroll
  for (int j = 0; j < 8; ++j) { run += loc[j]; op[t * 8 + j] = fmaxf(run, 1.0f); }
}

// ---------------- chunked linear-attention recurrence -----------------------

// A_c[f,d] = sum_{t in chunk} Ksh[t][f] * vg[t][d]   (64x256 per chunk)
__global__ __launch_bounds__(256) void chunk_state(const _Float16* __restrict__ KshT,
                                                   const _Float16* __restrict__ vgT,
                                                   float* __restrict__ Achunk) {
  int chunk = blockIdx.x;
  int w = threadIdx.x >> 5, lane = threadIdx.x & 31;
  long t0 = (long)chunk * CH;
  float* dst = Achunk + (size_t)chunk * FEAT * DD;
  for (int tile = w; tile < 64; tile += 8) {
    int ft = tile >> 4, dt = tile & 15;
    v8f acc = {};
#pragma unroll
    for (int kb = 0; kb < CH; kb += 32) {
      // A = Kshᵀ: (m=f, k=t) -> KshT[(ft*16+m)*BLT + t0+kb+k]  (contiguous k)
      v16h a = ld_frag(KshT + (size_t)(ft * 16) * BLT + t0 + kb, lane, BLT, 1);
      // B = vg:   (k=t, n=d) -> vgT[(dt*16+n)*BLT + t0+kb+k]   (contiguous k)
      v16h b = ld_frag(vgT + (size_t)(dt * 16) * BLT + t0 + kb, lane, BLT, 1);
      acc = wmma_f16(a, b, acc);
    }
    int n = lane & 15, ro = (lane & 16) ? 8 : 0;
#pragma unroll
    for (int v = 0; v < 8; ++v)
      dst[(size_t)(ft * 16 + v + ro) * DD + dt * 16 + n] = acc[v];
  }
}

// exclusive prefix of chunk states per batch, emitted transposed as f16:
// ShT[chunk][d][f]
__global__ __launch_bounds__(256) void chunk_prefix(const float* __restrict__ Achunk,
                                                    _Float16* __restrict__ ShT) {
  int b = blockIdx.x >> 6;
  int idx = (blockIdx.x & 63) * 256 + threadIdx.x;  // idx = f*256 + d
  int f = idx >> 8, d = idx & 255;
  float s = 0.f;
  for (int c = 0; c < LL / CH; ++c) {
    size_t gcid = (size_t)(b * (LL / CH) + c) * FEAT * DD;
    ShT[gcid + (size_t)d * FEAT + f] = (_Float16)s;
    s += Achunk[gcid + idx];
  }
}

// retr = (tril(Q Kshᵀ) vg + Q S_c) / (sqrt(gc)*sqrt(P))
__global__ __launch_bounds__(256) void chunk_retrieve(const _Float16* __restrict__ Q,
                                                      const _Float16* __restrict__ Ksh,
                                                      const _Float16* __restrict__ vgT,
                                                      const _Float16* __restrict__ ShT,
                                                      const float* __restrict__ gc,
                                                      float* __restrict__ retr) {
  int chunk = blockIdx.x;
  int w = threadIdx.x >> 5, lane = threadIdx.x & 31;
  long t0 = (long)chunk * CH;
  __shared__ _Float16 Msh[CH * CH];

  // Phase 1: causal-masked score matrix M[t][s] -> LDS (f16)
#pragma unroll
  for (int rep = 0; rep < 2; ++rep) {
    int tile = w + rep * 8;
    int mt1 = tile >> 2, st1 = tile & 3;
    v8f acc = {};
#pragma unroll
    for (int kb = 0; kb < FEAT; kb += 32) {
      // A = Q rows: (m=t, k=f), contiguous k
      v16h a = ld_frag(Q + (t0 + mt1 * 16) * FEAT + kb, lane, FEAT, 1);
      // B = Kshᵀ: (k=f, n=s) -> Ksh[(t0+st1*16+n)*FEAT + kb+k], contiguous k
      v16h b = ld_frag(Ksh + (t0 + st1 * 16) * FEAT + kb, lane, FEAT, 1);
      acc = wmma_f16(a, b, acc);
    }
    int sI = st1 * 16 + (lane & 15);
    int ro1 = (lane & 16) ? 8 : 0;
#pragma unroll
    for (int v = 0; v < 8; ++v) {
      int tI = mt1 * 16 + v + ro1;
      Msh[tI * CH + sI] = (_Float16)((sI <= tI) ? acc[v] : 0.f);
    }
  }
  __syncthreads();

  // Phase 2: R = M @ vg + Q @ S_c  (wave: 1 M-tile x 8 N-tiles)
  int mt = w & 3;
  int ngBase = (w >> 2) * 8;
  const _Float16* Sc = ShT + (size_t)chunk * FEAT * DD;  // [d][f]
  v16h aq0 = ld_frag(Q + (t0 + mt * 16) * FEAT + 0,  lane, FEAT, 1);
  v16h aq1 = ld_frag(Q + (t0 + mt * 16) * FEAT + 32, lane, FEAT, 1);
  v16h am0 = ld_frag(Msh + mt * 16 * CH + 0,  lane, CH, 1);
  v16h am1 = ld_frag(Msh + mt * 16 * CH + 32, lane, CH, 1);
  int n = lane & 15, ro = (lane & 16) ? 8 : 0;
  for (int nt = ngBase; nt < ngBase + 8; ++nt) {
    v8f acc = {};
    // B = S: (k=f, n=d) -> Sc[(nt*16+n)*FEAT + kb+k], contiguous k
    acc = wmma_f16(aq0, ld_frag(Sc + (size_t)(nt * 16) * FEAT + 0,  lane, FEAT, 1), acc);
    acc = wmma_f16(aq1, ld_frag(Sc + (size_t)(nt * 16) * FEAT + 32, lane, FEAT, 1), acc);
    // B = vg: (k=t, n=d) -> vgT[(nt*16+n)*BLT + t0+kb+k], contiguous k
    acc = wmma_f16(am0, ld_frag(vgT + (size_t)(nt * 16) * BLT + t0 + 0,  lane, BLT, 1), acc);
    acc = wmma_f16(am1, ld_frag(vgT + (size_t)(nt * 16) * BLT + t0 + 32, lane, BLT, 1), acc);
#pragma unroll
    for (int v = 0; v < 8; ++v) {
      long t = t0 + mt * 16 + v + ro;
      float sc = rsqrtf(gc[t]) * 0.17677669529663689f;  // 1/sqrt(P)
      retr[t * DD + nt * 16 + n] = acc[v] * sc;
    }
  }
}

// ---------------- LayerNorm / misc ------------------------------------------

__global__ __launch_bounds__(256) void ln_rows(const float* __restrict__ X,
                                               const float* __restrict__ bias,
                                               const float* __restrict__ gamma,
                                               const float* __restrict__ beta,
                                               _Float16* __restrict__ Yh) {
  int t = blockIdx.x, d = threadIdx.x;
  float x = X[(size_t)t * DD + d] + (bias ? bias[d] : 0.f);
  __shared__ float red[256];
  red[d] = x;
  __syncthreads();
#pragma unroll
  for (int s = 128; s > 0; s >>= 1) { if (d < s) red[d] += red[d + s]; __syncthreads(); }
  float m = red[0] * (1.0f / DD);
  __syncthreads();
  float c = x - m;
  red[d] = c * c;
  __syncthreads();
#pragma unroll
  for (int s = 128; s > 0; s >>= 1) { if (d < s) red[d] += red[d + s]; __syncthreads(); }
  float var = red[0] * (1.0f / DD);
  float y = c * rsqrtf(var + 1e-5f) * gamma[d] + beta[d];
  Yh[(size_t)t * DD + d] = (_Float16)y;
}

__global__ __launch_bounds__(256) void bias_gelu_f16(const float* __restrict__ Y,
                                                     const float* __restrict__ b,
                                                     _Float16* __restrict__ Oh,
                                                     int total, int Nn) {
  int i = blockIdx.x * 256 + threadIdx.x;
  if (i >= total) return;
  Oh[i] = (_Float16)gelu_exact(Y[i] + b[i % Nn]);
}

__global__ __launch_bounds__(256) void final_add(const float* __restrict__ x,
                                                 const float* __restrict__ Y4,
                                                 const float* __restrict__ bo,
                                                 float* __restrict__ out, int total) {
  int i = blockIdx.x * 256 + threadIdx.x;
  if (i >= total) return;
  out[i] = x[i] + Y4[i] + bo[i % DD];
}

// ---------------- launch ----------------------------------------------------

extern "C" void kernel_launch(void* const* d_in, const int* in_sizes, int n_in,
                              void* d_out, int out_size, void* d_ws, size_t ws_size,
                              hipStream_t stream) {
  const float* x    = (const float*)d_in[0];
  const float* Wk   = (const float*)d_in[1];
  const float* bk   = (const float*)d_in[2];
  const float* Wv   = (const float*)d_in[3];
  const float* bv   = (const float*)d_in[4];
  const float* Wg1  = (const float*)d_in[5];
  const float* bg1  = (const float*)d_in[6];
  const float* Wg2  = (const float*)d_in[7];
  const float* bg2  = (const float*)d_in[8];
  const float* ln1g = (const float*)d_in[9];
  const float* ln1b = (const float*)d_in[10];
  const float* Wr1  = (const float*)d_in[11];
  const float* br1  = (const float*)d_in[12];
  const float* Wr2  = (const float*)d_in[13];
  const float* br2  = (const float*)d_in[14];
  const float* ln2g = (const float*)d_in[15];
  const float* ln2b = (const float*)d_in[16];
  const float* Wo   = (const float*)d_in[17];
  const float* bo   = (const float*)d_in[18];
  float* out = (float*)d_out;

  char* wsb = (char*)d_ws;
  size_t off = 0;
  auto nxt = [&](size_t bytes) -> void* {
    void* p = wsb + off;
    off += (bytes + 255) & ~(size_t)255;
    return p;
  };

  const int N1 = PP + 3 * DD;  // 800: [Wk | Wv | Wg1_top | Wg1_bot]
  _Float16* xh    = (_Float16*)nxt((size_t)BLT * DD * 2);
  _Float16* WcatT = (_Float16*)nxt((size_t)N1 * DD * 2);      // [800, 256]
  _Float16* Wr1T  = (_Float16*)nxt((size_t)2 * DD * DD * 2);  // [512, 256]
  _Float16* Wr2T  = (_Float16*)nxt((size_t)DD * 2 * DD * 2);  // [256, 512]
  _Float16* WoT   = (_Float16*)nxt((size_t)DD * DD * 2);      // [256, 256]
  _Float16* Qf    = (_Float16*)nxt((size_t)BLT * FEAT * 2);
  _Float16* Kshf  = (_Float16*)nxt((size_t)BLT * FEAT * 2);   // [t][f]
  _Float16* KshT  = (_Float16*)nxt((size_t)FEAT * BLT * 2);   // [f][t]
  float*    gbuf  = (float*)nxt((size_t)BLT * 4);
  float*    gcb   = (float*)nxt((size_t)BLT * 4);
  _Float16* vgT   = (_Float16*)nxt((size_t)DD * BLT * 2);     // [d][t]
  float*    Ach   = (float*)nxt((size_t)NC * FEAT * DD * 4);
  _Float16* ShT   = (_Float16*)nxt((size_t)NC * FEAT * DD * 2);  // [c][d][f]
  float*    retr  = (float*)nxt((size_t)BLT * DD * 4);
  _Float16* hbuf  = (_Float16*)nxt((size_t)BLT * DD * 2);
  _Float16* R1h   = (_Float16*)nxt((size_t)BLT * 2 * DD * 2);
  _Float16* zh    = (_Float16*)nxt((size_t)BLT * DD * 2);
  float*    Ybig  = (float*)nxt((size_t)BLT * N1 * 4);  // reused for Y1/Y2/Y3/Y4

  auto gEl = [](size_t n) { return dim3((unsigned)((n + 255) / 256)); };
  auto gWv = [](int waves) { return dim3((unsigned)((waves + 7) / 8)); };

  // ---- convert inputs; build fused transposed weight [800 x 256] ----
  cvt_block<<<gEl((size_t)BLT * DD), 256, 0, stream>>>(x, xh, BLT, DD, DD);
  cvt_t<<<gEl((size_t)DD * PP), 256, 0, stream>>>(Wk, WcatT + (size_t)0 * DD, DD, PP, DD);
  cvt_t<<<gEl((size_t)DD * DD), 256, 0, stream>>>(Wv, WcatT + (size_t)PP * DD, DD, DD, DD);
  cvt_t<<<gEl((size_t)DD * DD), 256, 0, stream>>>(Wg1, WcatT + (size_t)(PP + DD) * DD,
                                                  DD, DD, DD);
  cvt_t<<<gEl((size_t)DD * DD), 256, 0, stream>>>(Wg1 + (size_t)DD * DD,
                                                  WcatT + (size_t)(PP + 2 * DD) * DD,
                                                  DD, DD, DD);
  cvt_t<<<gEl((size_t)DD * 2 * DD), 256, 0, stream>>>(Wr1, Wr1T, DD, 2 * DD, DD);
  cvt_t<<<gEl((size_t)2 * DD * DD), 256, 0, stream>>>(Wr2, Wr2T, 2 * DD, DD, 2 * DD);
  cvt_t<<<gEl((size_t)DD * DD), 256, 0, stream>>>(Wo, WoT, DD, DD, DD);

  // ---- fused input GEMM: Y1 = xh @ Wcat  [BLT x 800] ----
  gemm_f16f32<<<gWv((BLT / 16) * (N1 / 32)), 256, 0, stream>>>(xh, WcatT, Ybig,
                                                               BLT, N1, DD);

  // ---- epilogues: phasors, gate, shifted keys, gated values ----
  phase_kernel<<<BLT, 32, 0, stream>>>(Ybig, bk, Qf, N1);
  gate_kernel<<<BLT, 256, 0, stream>>>(Ybig, bv, bg1, Wg2, bg2, Qf, gbuf,
                                       vgT, Kshf, KshT, N1);
  gcum_kernel<<<BB, 256, 0, stream>>>(gbuf, gcb);

  // ---- chunked linear-attention recurrence ----
  chunk_state<<<NC, 256, 0, stream>>>(KshT, vgT, Ach);
  chunk_prefix<<<BB * 64, 256, 0, stream>>>(Ach, ShT);
  chunk_retrieve<<<NC, 256, 0, stream>>>(Qf, Kshf, vgT, ShT, gcb, retr);

  // ---- LN1 -> refine MLP -> LN2 -> output projection -> residual ----
  ln_rows<<<BLT, 256, 0, stream>>>(retr, nullptr, ln1g, ln1b, hbuf);
  gemm_f16f32<<<gWv((BLT / 16) * (2 * DD / 32)), 256, 0, stream>>>(hbuf, Wr1T, Ybig,
                                                                   BLT, 2 * DD, DD);
  bias_gelu_f16<<<gEl((size_t)BLT * 2 * DD), 256, 0, stream>>>(Ybig, br1, R1h,
                                                               BLT * 2 * DD, 2 * DD);
  gemm_f16f32<<<gWv((BLT / 16) * (DD / 32)), 256, 0, stream>>>(R1h, Wr2T, Ybig,
                                                               BLT, DD, 2 * DD);
  ln_rows<<<BLT, 256, 0, stream>>>(Ybig, br2, ln2g, ln2b, zh);
  gemm_f16f32<<<gWv((BLT / 16) * (DD / 32)), 256, 0, stream>>>(zh, WoT, Ybig,
                                                               BLT, DD, DD);
  final_add<<<gEl((size_t)BLT * DD), 256, 0, stream>>>(x, Ybig, bo, out, BLT * DD);
}
